// ToyModel_68547678044622
// MI455X (gfx1250) — 8-bit matrix operand order repaired and checked by reading, compiled, not run on hardware
//
#include <hip/hip_runtime.h>
#include <hip/hip_bf16.h>


typedef int v8i __attribute__((ext_vector_type(8)));

#define EPS 1e-5f
#define BATCH 256
#define H 4096
#define HO1 4094
#define HO2 4092

__global__ void k1_conv1_bn_binpack(const float* __restrict__ x,
                                    const float* __restrict__ w1,
                                    const float* __restrict__ g1, const float* __restrict__ b1,
                                    const float* __restrict__ m1, const float* __restrict__ v1,
                                    unsigned int* __restrict__ s1bits)
{
    __shared__ float c0s[32], c1s[32], c2s[32], ds[32];
    int t = threadIdx.x;
    if (t < 32) {
        float a  = g1[t] * rsqrtf(v1[t] + EPS);
        float w0 = w1[t*3+0], wA = w1[t*3+1], w2v = w1[t*3+2];
        float s0 = (w0 > 0.f) ? 1.f : ((w0 < 0.f) ? -1.f : 0.f);
        float s1 = (wA > 0.f) ? 1.f : ((wA < 0.f) ? -1.f : 0.f);
        float s2 = (w2v > 0.f) ? 1.f : ((w2v < 0.f) ? -1.f : 0.f);
        c0s[t] = a * s0; c1s[t] = a * s1; c2s[t] = a * s2;
        ds[t]  = b1[t] - m1[t] * a;
    }
    __syncthreads();
    long long p = (long long)blockIdx.x * blockDim.x + t;
    if (p >= (long long)BATCH * HO1) return;
    int b = (int)(p / HO1);
    int i = (int)(p % HO1);
    const float* xb = x + (long long)b * H + i;
    float x0 = xb[0], x1 = xb[1], x2 = xb[2];
    unsigned int bits = 0u;
    #pragma unroll
    for (int c = 0; c < 32; ++c) {
        float y = c0s[c]*x0 + c1s[c]*x1 + c2s[c]*x2 + ds[c];
        bits |= (y > 0.f ? 1u : 0u) << c;
    }
    s1bits[p] = bits;
}

__global__ void k2_conv2_bn_bin(const unsigned int* __restrict__ s1bits,
                                const float* __restrict__ w2,
                                const float* __restrict__ g2, const float* __restrict__ b2,
                                const float* __restrict__ m2, const float* __restrict__ v2,
                                signed char* __restrict__ s2)
{
    __shared__ unsigned int wb[32][3];
    __shared__ float a2s[32], d2s[32];
    int t = threadIdx.x;
    if (t < 32) {
        float a = g2[t] * rsqrtf(v2[t] + EPS);
        a2s[t] = a; d2s[t] = b2[t] - m2[t]*a;
        unsigned int bk0=0, bk1=0, bk2=0;
        for (int c = 0; c < 32; ++c) {
            const float* wp = w2 + ((long long)t*32 + c)*3;
            bk0 |= (wp[0] > 0.f ? 1u : 0u) << c;
            bk1 |= (wp[1] > 0.f ? 1u : 0u) << c;
            bk2 |= (wp[2] > 0.f ? 1u : 0u) << c;
        }
        wb[t][0]=bk0; wb[t][1]=bk1; wb[t][2]=bk2;
    }
    __syncthreads();
    const int Q = HO2 / 4;
    long long gid = (long long)blockIdx.x * blockDim.x + t;
    if (gid >= (long long)BATCH * 32 * Q) return;
    int q = (int)(gid % Q);
    long long bo = gid / Q;
    int o = (int)(bo % 32);
    int b = (int)(bo / 32);
    int i0 = q * 4;
    const unsigned int* sp = s1bits + (long long)b * HO1 + i0;
    unsigned int sv[6];
    #pragma unroll
    for (int j = 0; j < 6; ++j) sv[j] = sp[j];
    unsigned int u0 = wb[o][0], u1 = wb[o][1], u2 = wb[o][2];
    float a = a2s[o], d = d2s[o];
    int out4 = 0;
    #pragma unroll
    for (int j = 0; j < 4; ++j) {
        int dot = 96 - 2*(__popc(sv[j]^u0) + __popc(sv[j+1]^u1) + __popc(sv[j+2]^u2));
        float y = a * (float)dot + d;
        int sb = (y > 0.f) ? 1 : -1;
        out4 |= (sb & 0xFF) << (8*j);
    }
    *(int*)(s2 + ((long long)(b*32 + o)*HO2 + i0)) = out4;
}

__global__ void k3_fc_wmma(const signed char* __restrict__ s2,
                           const float* __restrict__ wf,
                           const float* __restrict__ g3, const float* __restrict__ b3,
                           const float* __restrict__ m3, const float* __restrict__ v3,
                           float* __restrict__ out, long long ntiles, int tiles_per_wave)
{
    const int lane = threadIdx.x & 31;
    const int wave = threadIdx.x >> 5;
    const int n  = lane & 15;
    const int hi = lane >> 4;

    v8i B0 = {}, B1 = {};
    #pragma unroll
    for (int ot = 0; ot < 2; ++ot) {
        int o = ot*16 + n;
        #pragma unroll
        for (int vg = 0; vg < 4; ++vg) {
            unsigned int w = 0;
            #pragma unroll
            for (int by = 0; by < 4; ++by) {
                int j = hi*16 + vg*4 + by;
                float wv = wf[o*32 + j];
                int sb = (wv > 0.f) ? 1 : ((wv < 0.f) ? -1 : 0);
                w |= ((unsigned)(sb & 0xFF)) << (8*by);
            }
            if (ot == 0) B0[vg] = (int)w; else B1[vg] = (int)w;
        }
    }
    const int o0 = n, o1 = 16 + n;
    float a30 = g3[o0] * rsqrtf(v3[o0] + EPS);
    float a31 = g3[o1] * rsqrtf(v3[o1] + EPS);
    float d30 = b3[o0] - m3[o0]*a30;
    float d31 = b3[o1] - m3[o1]*a31;

    long long wgid  = (long long)blockIdx.x * (blockDim.x >> 5) + wave;
    long long tile0 = wgid * tiles_per_wave;
    for (int tt = 0; tt < tiles_per_wave; ++tt) {
        long long tile = tile0 + tt;
        if (tile >= ntiles) break;
        const signed char* arow = s2 + tile*512 + (long long)n*32;
        long long ptile = (tile + 1 < ntiles) ? tile + 1 : tile;
        __builtin_prefetch(s2 + ptile*512 + (long long)n*32, 0, 1);

        int2 l0 = *(const int2*)(arow + hi*16);
        int2 l1 = *(const int2*)(arow + hi*16 + 8);
        v8i A = {};
        A[0] = l0.x; A[1] = l0.y; A[2] = l1.x; A[3] = l1.y;

        v8i C = {};
        v8i D0 = __builtin_amdgcn_wmma_i32_16x16x64_iu8(true, A, true, B0, C, false, false);
        v8i D1 = __builtin_amdgcn_wmma_i32_16x16x64_iu8(true, A, true, B1, C, false, false);

        float* obase = out + tile * (16*32);
        #pragma unroll
        for (int v = 0; v < 8; ++v) {
            int m = v + 8*hi;
            obase[m*32 + n]      = a30 * (float)D0[v] + d30;
            obase[m*32 + 16 + n] = a31 * (float)D1[v] + d31;
        }
    }
}

extern "C" void kernel_launch(void* const* d_in, const int* in_sizes, int n_in,
                              void* d_out, int out_size, void* d_ws, size_t ws_size,
                              hipStream_t stream) {
    const float* x  = (const float*)d_in[0];
    const float* w1 = (const float*)d_in[1];
    const float* w2 = (const float*)d_in[2];
    const float* wf = (const float*)d_in[3];
    const float* g1 = (const float*)d_in[4];
    const float* b1 = (const float*)d_in[5];
    const float* m1 = (const float*)d_in[6];
    const float* v1 = (const float*)d_in[7];
    const float* g2 = (const float*)d_in[8];
    const float* b2 = (const float*)d_in[9];
    const float* m2 = (const float*)d_in[10];
    const float* v2 = (const float*)d_in[11];
    const float* g3 = (const float*)d_in[12];
    const float* b3 = (const float*)d_in[13];
    const float* m3 = (const float*)d_in[14];
    const float* v3 = (const float*)d_in[15];
    float* out = (float*)d_out;

    unsigned int* s1bits = (unsigned int*)d_ws;
    const size_t s1_bytes = (size_t)BATCH * HO1 * sizeof(unsigned int);
    signed char* s2 = (signed char*)d_ws + s1_bytes;

    {
        long long total = (long long)BATCH * HO1;
        int blocks = (int)((total + 255) / 256);
        k1_conv1_bn_binpack<<<blocks, 256, 0, stream>>>(x, w1, g1, b1, m1, v1, s1bits);
    }
    {
        long long total = (long long)BATCH * 32 * (HO2 / 4);
        int blocks = (int)((total + 255) / 256);
        k2_conv2_bn_bin<<<blocks, 256, 0, stream>>>(s1bits, w2, g2, b2, m2, v2, s2);
    }
    {
        long long nrows  = (long long)BATCH * HO2;
        long long ntiles = nrows / 16;
        const int tiles_per_wave = 8;
        const int waves_per_block = 8;
        long long tiles_per_block = (long long)tiles_per_wave * waves_per_block;
        int blocks = (int)((ntiles + tiles_per_block - 1) / tiles_per_block);
        k3_fc_wmma<<<blocks, 256, 0, stream>>>(s2, wf, g3, b3, m3, v3, out,
                                               ntiles, tiles_per_wave);
    }
}
